// IASSDHead_65876208386767
// MI455X (gfx1250) — matrix operand order repaired and checked by reading, not compiled, not run on hardware
//
#include <hip/hip_runtime.h>
#include <hip/hip_bf16.h>
#include <math.h>

typedef __attribute__((ext_vector_type(16))) _Float16 v16h;
typedef __attribute__((ext_vector_type(8)))  _Float16 v8h;
typedef __attribute__((ext_vector_type(8)))  float    v8f;

constexpr int Bb   = 8;
constexpr int Nn   = 4096;
constexpr int Cc   = 512;
constexpr int MIDc = 256;
constexpr int CODE = 30;   // 6 + 2*NBIN
constexpr int MBOX = 64;
constexpr float PI_F  = 3.14159265358979323846f;
constexpr float BINW  = 2.0f * PI_F / 12.0f;

// workspace layout (f16 elements)
constexpr int WS_W1B = 0;                       // 256*512
constexpr int WS_W1C = WS_W1B + MIDc * Cc;      // 256*512
constexpr int WS_W2B = WS_W1C + MIDc * Cc;      // 32*256 (rows >=30 zero)
constexpr int WS_W2C = WS_W2B + 32 * MIDc;      // 16*256 (rows >=3 zero)
constexpr int WS_TOT = WS_W2C + 16 * MIDc;      // 274432 f16 = 548864 B

// ---------------------------------------------------------------------------
// One-shot weight pack: fp32 -> f16, W2 zero-padded to 16-row tiles.
// ---------------------------------------------------------------------------
__global__ void pack_weights_kernel(const float* __restrict__ Wb1,
                                    const float* __restrict__ Wc1,
                                    const float* __restrict__ Wb2,
                                    const float* __restrict__ Wc2,
                                    _Float16* __restrict__ ws)
{
    const int t = blockIdx.x * blockDim.x + threadIdx.x;
    if (t < WS_W1C) {
        ws[t] = (_Float16)Wb1[t - WS_W1B];
    } else if (t < WS_W2B) {
        ws[t] = (_Float16)Wc1[t - WS_W1C];
    } else if (t < WS_W2C) {
        const int i = t - WS_W2B, row = i / MIDc, col = i % MIDc;
        ws[t] = (row < 30) ? (_Float16)Wb2[row * MIDc + col] : (_Float16)0.0f;
    } else if (t < WS_TOT) {
        const int i = t - WS_W2C, row = i / MIDc, col = i % MIDc;
        ws[t] = (row < 3) ? (_Float16)Wc2[row * MIDc + col] : (_Float16)0.0f;
    }
}

// A-fragment loader: ISA 16-bit A layout — lane holds row m, K = 8h+(i&7)
// (+16 for i>=8) => two contiguous, 16B-aligned groups of 8 f16.
__device__ __forceinline__ v16h load_afrag(const _Float16* __restrict__ p)
{
    const v8h lo = *(const v8h*)p;
    const v8h hi = *(const v8h*)(p + 16);
    return __builtin_shufflevector(lo, hi, 0, 1, 2, 3, 4, 5, 6, 7,
                                           8, 9, 10, 11, 12, 13, 14, 15);
}

// ---------------------------------------------------------------------------
// Fused 2-layer MLP head (both heads), WMMA f16 inputs -> f32 accumulate.
// One wave owns a 16-column tile of N. GEMM1 (256x16 over C=512) -> LDS f16,
// GEMM2 (padded W2 @ H) -> transposed (B,N,code) outputs.
// ---------------------------------------------------------------------------
__global__ void __launch_bounds__(128)
head_gemm_kernel(const float* __restrict__ x,
                 const _Float16* __restrict__ wsw,
                 const float* __restrict__ bb1, const float* __restrict__ sb1,
                 const float* __restrict__ tb1, const float* __restrict__ bb2,
                 const float* __restrict__ bc1, const float* __restrict__ sc1,
                 const float* __restrict__ tc1, const float* __restrict__ bc2,
                 float* __restrict__ outBox, float* __restrict__ outCls)
{
    __shared__ _Float16 Hs[4][MIDc * 16];   // 32 KB: per-wave 256x16 f16 tile

    const int lane = threadIdx.x & 31;
    const int wave = threadIdx.x >> 5;
    const int m    = lane & 15;   // A-row / B-col / D-col index
    const int hh   = lane >> 4;   // half-wave selector
    const int tile = blockIdx.x * 4 + wave;
    const int b    = tile / (Nn / 16);
    const int n0   = (tile % (Nn / 16)) * 16;
    _Float16* hbuf = &Hs[wave][0];

    const v8f vzero = {};

    for (int hd = 0; hd < 2; ++hd) {
        const _Float16* W1 = wsw + (hd ? WS_W1C : WS_W1B);
        const _Float16* W2 = wsw + (hd ? WS_W2C : WS_W2B);
        const float* B1 = hd ? bc1 : bb1;
        const float* S1 = hd ? sc1 : sb1;
        const float* T1 = hd ? tc1 : tb1;
        const float* B2 = hd ? bc2 : bb2;
        const int vRows = hd ? 3 : 30;     // valid output rows
        const int nOT   = hd ? 1 : 2;      // 16-row output tiles
        const int outW  = hd ? 3 : 30;
        float* outp     = hd ? outCls : outBox;

        // ---- GEMM1: accumulate 256 x 16 over C=512 (16 K-chunks of 32) ----
        v8f acc[16];
        #pragma unroll
        for (int o = 0; o < 16; ++o) acc[o] = vzero;

        for (int cc = 0; cc < 16; ++cc) {
            const int c0 = cc * 32;
            // B fragment: x[c0+8h+(i&7)+16*(i>>3)][n0+m], f32 -> f16 (amortized over 16 WMMA)
            v16h bx;
            #pragma unroll
            for (int i = 0; i < 16; ++i)
                bx[i] = (_Float16)x[(b * Cc + c0 + hh * 8 + (i & 7) + ((i >> 3) << 4)) * Nn + n0 + m];

            const _Float16* wbase = W1 + m * Cc + c0 + hh * 8;
            #pragma unroll
            for (int o = 0; o < 16; ++o) {
                const v16h af = load_afrag(wbase + o * (16 * Cc));
                acc[o] = __builtin_amdgcn_wmma_f32_16x16x32_f16(
                    false, af, false, bx, (short)0, acc[o], false, false);
            }
        }

        // ---- epilogue: relu((acc + b1)*s1 + t1) -> LDS f16 ----
        #pragma unroll
        for (int o = 0; o < 16; ++o) {
            #pragma unroll
            for (int r = 0; r < 8; ++r) {
                const int row = o * 16 + r + 8 * hh;   // D: M = r + 8h
                float v = acc[o][r] + B1[row];
                v = v * S1[row] + T1[row];
                v = fmaxf(v, 0.0f);
                hbuf[row * 16 + m] = (_Float16)v;
            }
        }
        __syncthreads();

        // ---- GEMM2: (padded rows x 256) @ H(256x16) ----
        for (int ot = 0; ot < nOT; ++ot) {
            v8f acc2 = vzero;
            const _Float16* w2base = W2 + (ot * 16 + m) * MIDc + hh * 8;
            for (int k = 0; k < 8; ++k) {
                v16h bh;
                #pragma unroll
                for (int i = 0; i < 16; ++i)
                    bh[i] = hbuf[(k * 32 + hh * 8 + (i & 7) + ((i >> 3) << 4)) * 16 + m];
                const v16h aw = load_afrag(w2base + k * 32);
                acc2 = __builtin_amdgcn_wmma_f32_16x16x32_f16(
                    false, aw, false, bh, (short)0, acc2, false, false);
            }
            #pragma unroll
            for (int r = 0; r < 8; ++r) {
                const int orow = ot * 16 + r + 8 * hh;
                if (orow < vRows) {
                    const float v = acc2[r] + B2[orow];
                    outp[(b * Nn + n0 + m) * outW + orow] = v;  // (B,N,code)
                }
            }
        }
        __syncthreads();
    }
}

// ---------------------------------------------------------------------------
// decode: per point, argmax class, anchor math, bin/residual heading.
// ---------------------------------------------------------------------------
__global__ void decode_kernel(const float* __restrict__ boxp,
                              const float* __restrict__ clsp,
                              const float* __restrict__ points,
                              const float* __restrict__ mean_size,
                              float* __restrict__ outp)
{
    const int t = blockIdx.x * blockDim.x + threadIdx.x;
    if (t >= Bb * Nn) return;
    const float* p = boxp + t * CODE;
    const float* c = clsp + t * 3;

    int cls = 0; float best = c[0];
    #pragma unroll
    for (int j = 1; j < 3; ++j) { float v = c[j]; if (v > best) { best = v; cls = j; } }

    const float a0 = mean_size[cls * 3 + 0];
    const float a1 = mean_size[cls * 3 + 1];
    const float a2 = mean_size[cls * 3 + 2];
    const float diag = sqrtf(a0 * a0 + a1 * a1);
    const float sc[3]  = {diag, diag, a2};
    const float anc[3] = {a0, a1, a2};

    float ov[7];
    #pragma unroll
    for (int j = 0; j < 3; ++j) ov[j]     = points[t * 3 + j] + p[j] * sc[j];
    #pragma unroll
    for (int j = 0; j < 3; ++j) ov[3 + j] = expf(p[3 + j]) * anc[j];

    int bbin = 0; float bl = p[6];
    #pragma unroll
    for (int j = 1; j < 12; ++j) { float v = p[6 + j]; if (v > bl) { bl = v; bbin = j; } }
    const float r = p[6 + 12 + bbin];
    ov[6] = ((float)bbin + 0.5f) * BINW + r * (BINW * 0.5f) - PI_F;

    #pragma unroll
    for (int j = 0; j < 7; ++j) outp[t * 7 + j] = ov[j];
}

// ---------------------------------------------------------------------------
// assign: points-in-(boxes / ext-boxes), label gather, optional encode.
// Blocks never straddle a batch (npts % 256 == 0), boxes staged in LDS.
// ---------------------------------------------------------------------------
__global__ void assign_kernel(const float* __restrict__ pts, int npts,
                              const float* __restrict__ gt_boxes,
                              const int*   __restrict__ gt_labels,
                              const float* __restrict__ ext,
                              int ignoreFlag, int retLabels,
                              const float* __restrict__ mean_size,
                              float* __restrict__ out_idx,
                              float* __restrict__ out_cls,
                              float* __restrict__ out_fgb,
                              float* __restrict__ out_lab)
{
    __shared__ float sbox[MBOX * 7];
    const int t  = blockIdx.x * blockDim.x + threadIdx.x;
    const int b  = t / npts;
    const int bb = (blockIdx.x * (int)blockDim.x) / npts;   // block-uniform batch
    for (int i = threadIdx.x; i < MBOX * 7; i += blockDim.x)
        sbox[i] = gt_boxes[bb * MBOX * 7 + i];
    __syncthreads();

    const float ex = ext[0], ey = ext[1], ez = ext[2];
    const float px = pts[t * 3 + 0], py = pts[t * 3 + 1], pz = pts[t * 3 + 2];

    int idx = -1, eidx = -1;
    for (int mN = 0; mN < MBOX; ++mN) {
        const float* bx = sbox + mN * 7;
        const float rx = px - bx[0], ry = py - bx[1], rz = pz - bx[2];
        const float ch = cosf(bx[6]), sh = sinf(bx[6]);
        const float lx =  rx * ch + ry * sh;
        const float ly = -rx * sh + ry * ch;
        const float ax = fabsf(lx), ay = fabsf(ly), az = fabsf(rz);
        const bool in0 = (ax <= bx[3] * 0.5f) && (ay <= bx[4] * 0.5f) && (az <= bx[5] * 0.5f);
        const bool in1 = (ax <= (bx[3] + ex) * 0.5f) && (ay <= (bx[4] + ey) * 0.5f)
                      && (az <= (bx[5] + ez) * 0.5f);
        if (in0 && idx  < 0) idx  = mN;   // first-inside == jnp.argmax semantics
        if (in1 && eidx < 0) eidx = mN;
    }

    bool fg = idx >= 0;
    const bool efg = eidx >= 0;
    float clsv; int oIdx;
    if (ignoreFlag) { clsv = (fg != efg) ? -1.0f : 0.0f; oIdx = idx; }
    else            { oIdx = fg ? idx : eidx; fg = efg;  clsv = 0.0f; }

    const int idx_c = oIdx < 0 ? 0 : oIdx;
    const int lab   = gt_labels[b * MBOX + idx_c];
    if (fg) clsv = (float)lab;
    fg = fg && (clsv != 0.0f);

    out_idx[t] = (float)oIdx;
    out_cls[t] = clsv;
    const float* gb = sbox + idx_c * 7;
    #pragma unroll
    for (int j = 0; j < 7; ++j) out_fgb[t * 7 + j] = fg ? gb[j] : 0.0f;

    if (retLabels) {
        const int lc = lab < 1 ? 1 : (lab > 3 ? 3 : lab);
        const float a0 = mean_size[(lc - 1) * 3 + 0];
        const float a1 = mean_size[(lc - 1) * 3 + 1];
        const float a2 = mean_size[(lc - 1) * 3 + 2];
        const float diag = sqrtf(a0 * a0 + a1 * a1);
        float enc[8];
        enc[0] = (gb[0] - px) / diag;
        enc[1] = (gb[1] - py) / diag;
        enc[2] = (gb[2] - pz) / a2;
        enc[3] = logf(fmaxf(gb[3], 0.001f) / a0);
        enc[4] = logf(fmaxf(gb[4], 0.001f) / a1);
        enc[5] = logf(fmaxf(gb[5], 0.001f) / a2);
        float ang = gb[6] + PI_F;
        ang = ang - 2.0f * PI_F * floorf(ang / (2.0f * PI_F));   // jnp.mod
        float bf = floorf(ang / BINW);
        bf = fminf(fmaxf(bf, 0.0f), 11.0f);
        enc[6] = bf;
        enc[7] = (ang - (bf + 0.5f) * BINW) / (BINW * 0.5f);
        #pragma unroll
        for (int j = 0; j < 8; ++j) out_lab[t * 8 + j] = fg ? enc[j] : 0.0f;
    }
}

// ---------------------------------------------------------------------------
extern "C" void kernel_launch(void* const* d_in, const int* in_sizes, int n_in,
                              void* d_out, int out_size, void* d_ws, size_t ws_size,
                              hipStream_t stream)
{
    (void)in_sizes; (void)n_in; (void)out_size; (void)ws_size;

    const float* ctr_feats   = (const float*)d_in[0];
    const float* ctr_preds   = (const float*)d_in[1];
    const float* ctr_origins = (const float*)d_in[2];
    const float* sa_pts0     = (const float*)d_in[3];
    const float* sa_pts1     = (const float*)d_in[4];
    const float* gt_boxes    = (const float*)d_in[5];
    const int*   gt_labels   = (const int*)d_in[6];
    const float* mean_size   = (const float*)d_in[7];
    const float* gt_ext      = (const float*)d_in[8];
    const float* sa_ext      = (const float*)d_in[9];
    const float* org_ext     = (const float*)d_in[10];
    const float* Wb1 = (const float*)d_in[11];
    const float* bb1 = (const float*)d_in[12];
    const float* sb1 = (const float*)d_in[13];
    const float* tb1 = (const float*)d_in[14];
    const float* Wb2 = (const float*)d_in[15];
    const float* bb2 = (const float*)d_in[16];
    const float* Wc1 = (const float*)d_in[17];
    const float* bc1 = (const float*)d_in[18];
    const float* sc1 = (const float*)d_in[19];
    const float* tc1 = (const float*)d_in[20];
    const float* Wc2 = (const float*)d_in[21];
    const float* bc2 = (const float*)d_in[22];
    float* out = (float*)d_out;
    _Float16* wsw = (_Float16*)d_ws;

    constexpr int NSA0 = 16384;
    // flat output offsets, reference return order
    constexpr int O_BOX = 0;
    constexpr int O_CLS = O_BOX + Bb * Nn * CODE;
    constexpr int O_PBP = O_CLS + Bb * Nn * 3;
    constexpr int O_CI  = O_PBP + Bb * Nn * 7;
    constexpr int O_CC  = O_CI  + Bb * Nn;
    constexpr int O_CF  = O_CC  + Bb * Nn;
    constexpr int O_CL  = O_CF  + Bb * Nn * 7;
    constexpr int O_OI  = O_CL  + Bb * Nn * 8;
    constexpr int O_OC  = O_OI  + Bb * Nn;
    constexpr int O_OF  = O_OC  + Bb * Nn;
    constexpr int O_OL  = O_OF  + Bb * Nn * 7;
    constexpr int O_S0I = O_OL  + Bb * Nn * 8;
    constexpr int O_S0C = O_S0I + Bb * NSA0;
    constexpr int O_S0F = O_S0C + Bb * NSA0;
    constexpr int O_S1I = O_S0F + Bb * NSA0 * 7;
    constexpr int O_S1C = O_S1I + Bb * Nn;
    constexpr int O_S1F = O_S1C + Bb * Nn;

    pack_weights_kernel<<<(WS_TOT + 255) / 256, 256, 0, stream>>>(
        Wb1, Wc1, Wb2, Wc2, wsw);

    head_gemm_kernel<<<(Bb * Nn / 16) / 4, 128, 0, stream>>>(
        ctr_feats, wsw, bb1, sb1, tb1, bb2, bc1, sc1, tc1, bc2,
        out + O_BOX, out + O_CLS);

    decode_kernel<<<(Bb * Nn) / 256, 256, 0, stream>>>(
        out + O_BOX, out + O_CLS, ctr_preds, mean_size, out + O_PBP);

    assign_kernel<<<(Bb * Nn) / 256, 256, 0, stream>>>(
        ctr_preds, Nn, gt_boxes, gt_labels, gt_ext, 1, 1, mean_size,
        out + O_CI, out + O_CC, out + O_CF, out + O_CL);

    assign_kernel<<<(Bb * Nn) / 256, 256, 0, stream>>>(
        ctr_origins, Nn, gt_boxes, gt_labels, org_ext, 0, 1, mean_size,
        out + O_OI, out + O_OC, out + O_OF, out + O_OL);

    assign_kernel<<<(Bb * NSA0) / 256, 256, 0, stream>>>(
        sa_pts0, NSA0, gt_boxes, gt_labels, sa_ext, 1, 0, mean_size,
        out + O_S0I, out + O_S0C, out + O_S0F, nullptr);

    assign_kernel<<<(Bb * Nn) / 256, 256, 0, stream>>>(
        sa_pts1, Nn, gt_boxes, gt_labels, sa_ext, 0, 0, mean_size,
        out + O_S1I, out + O_S1C, out + O_S1F, nullptr);
}
